// MyMSA_22574348108546
// MI455X (gfx1250) — matrix operand order repaired and checked by reading, not compiled, not run on hardware
//
#include <hip/hip_runtime.h>

// Multi-head self-attention, B=4, N=2048, D=256, H=8, DH=32.
// Stage 1: QKV projection (fp32 math), emit f16 Q (scaled by log2e/sqrt(DH)),
//          f16 K row-major, f16 V transposed ([DH][N]) into workspace.
// Stage 2: flash-attention per (b,h,16-query tile) using v_wmma_f32_16x16x32_f16:
//          S^T = K * Q^T  (lane owns one query column -> cheap online softmax)
//          O^T += V^T * P (P rebuilt in B-layout with a 4-dword lane-pair swap)
// Softmax runs in the base-2 domain (log2e folded into Q) and uses the raw
// hardware v_exp_f32 via __builtin_amdgcn_exp2f.

#define Bb 4
#define Nn 2048
#define Dd 256
#define Hh 8
#define DHh 32

typedef __attribute__((ext_vector_type(16))) _Float16 v16h;
typedef __attribute__((ext_vector_type(2)))  _Float16 h2;
typedef __attribute__((ext_vector_type(8)))  float    v8f;

// ---------------- Stage 1: QKV projection ----------------
__global__ __launch_bounds__(256) void qkv_proj(
    const float* __restrict__ x,
    const float* __restrict__ Wq, const float* __restrict__ bq,
    const float* __restrict__ Wk, const float* __restrict__ bk,
    const float* __restrict__ Wv, const float* __restrict__ bv,
    _Float16* __restrict__ Qh, _Float16* __restrict__ Kh,
    _Float16* __restrict__ Vt)
{
    int idx = blockIdx.x * blockDim.x + threadIdx.x;   // B*N*H*DH threads, e fastest
    int e = idx & (DHh - 1);
    int t = idx >> 5;
    int h = t & (Hh - 1);
    t >>= 3;
    int n = t & (Nn - 1);
    int b = t >> 11;

    const float* xr = x + (((size_t)b * Nn + n) * Dd + h * DHh);
    const float* wq = Wq + (size_t)(h * DHh + e) * DHh;
    const float* wk = Wk + (size_t)(h * DHh + e) * DHh;
    const float* wv = Wv + (size_t)(h * DHh + e) * DHh;

    float q = bq[h * DHh + e];
    float k = bk[h * DHh + e];
    float v = bv[h * DHh + e];
#pragma unroll
    for (int d = 0; d < DHh; ++d) {
        float xv = xr[d];
        q = fmaf(xv, wq[d], q);
        k = fmaf(xv, wk[d], k);
        v = fmaf(xv, wv[d], v);
    }
    // Fold softmax scale (1/sqrt(DH)) AND log2(e) into Q so stage 2 can use exp2.
    const float QSCALE = 1.4426950408889634f / 5.656854249492380195f;
    size_t row = ((size_t)(b * Hh + h) * Nn + n) * DHh + e;
    Qh[row] = (_Float16)(q * QSCALE);
    Kh[row] = (_Float16)k;
    Vt[((size_t)(b * Hh + h) * DHh + e) * Nn + n] = (_Float16)v;
}

// ---------------- Stage 2: attention ----------------
__global__ __launch_bounds__(128) void attn(
    const _Float16* __restrict__ Qh, const _Float16* __restrict__ Kh,
    const _Float16* __restrict__ Vt, float* __restrict__ out)
{
    const int lane = threadIdx.x & 31;
    const int wave = threadIdx.x >> 5;
    const int col  = lane & 15;        // query column (S^T N-index) / matrix row M
    const int hi   = (lane >> 4) & 1;  // lane half select (WMMA layout)

    int tl    = blockIdx.x * 4 + wave;        // 0..4095
    int qtile = tl & (Nn / 16 - 1);           // 0..127
    int bh    = tl >> 7;                      // b*H + h
    int h     = bh & (Hh - 1);
    int b     = bh >> 3;

    const _Float16* qbase = Qh + ((size_t)bh * Nn + qtile * 16) * DHh;
    const _Float16* kbase = Kh + (size_t)bh * Nn * DHh;
    const _Float16* vbase = Vt + (size_t)bh * DHh * Nn;

    // B operand for S^T = K * Q^T : lane holds Q[qbase+col][16*hi .. 16*hi+15]
    v16h bQ;
    {
        const _Float16* qr = qbase + col * DHh + 16 * hi;
#pragma unroll
        for (int j = 0; j < 16; ++j) bQ[j] = qr[j];
    }

    v8f acc0 = {};      // out^T tile d=0..15
    v8f acc1 = {};      // out^T tile d=16..31
    const v8f z = {};
    float mrun = -__builtin_inff();
    float lsum = 0.0f;

    for (int c = 0; c < Nn / 32; ++c) {
        const int m0 = c * 32;

        // A operand (K rows), 16-bit A layout: K = j + 16*hi (contiguous)
        v16h aK0, aK1;
        {
            const _Float16* k0 = kbase + (size_t)(m0 + col) * DHh;
            const _Float16* k1 = kbase + (size_t)(m0 + 16 + col) * DHh;
#pragma unroll
            for (int j = 0; j < 8; ++j) {
                aK0[j]     = k0[16 * hi + j];
                aK0[j + 8] = k0[16 * hi + 8 + j];
                aK1[j]     = k1[16 * hi + j];
                aK1[j + 8] = k1[16 * hi + 8 + j];
            }
        }
        v8f s0 = __builtin_amdgcn_wmma_f32_16x16x32_f16(false, aK0, false, bQ,
                                                        (short)0, z, false, false);
        v8f s1 = __builtin_amdgcn_wmma_f32_16x16x32_f16(false, aK1, false, bQ,
                                                        (short)0, z, false, false);
        // lane holds S^T values: m = m0 + r + 8*hi (s0) and m0 + 16 + r + 8*hi (s1)

        // --- online softmax (base-2 domain) ---
        float mx = fmaxf(s0[0], s1[0]);
#pragma unroll
        for (int r = 1; r < 8; ++r) mx = fmaxf(mx, fmaxf(s0[r], s1[r]));
        mx = fmaxf(mx, __shfl_xor(mx, 16, 32));   // combine lane-pair halves
        float mnew = fmaxf(mrun, mx);
        float fac  = __builtin_amdgcn_exp2f(mrun - mnew);  // first iter: 0
        mrun = mnew;

        float p0[8], p1[8], ps = 0.0f;
#pragma unroll
        for (int r = 0; r < 8; ++r) {
            p0[r] = __builtin_amdgcn_exp2f(s0[r] - mnew);   // raw v_exp_f32
            p1[r] = __builtin_amdgcn_exp2f(s1[r] - mnew);
            ps += p0[r] + p1[r];
        }
        lsum = fmaf(lsum, fac, ps);

        // --- rebuild P in B layout: bP half j <- P[m0 + 16*hi + j][col] ---
        // pack to f16 dwords, swap 4 dwords with partner lane (xor 16)
        int Ad[4], Bd[4];
#pragma unroll
        for (int r = 0; r < 4; ++r) {
            Ad[r] = __builtin_bit_cast(int, __builtin_amdgcn_cvt_pkrtz(p0[2*r], p0[2*r+1]));
            Bd[r] = __builtin_bit_cast(int, __builtin_amdgcn_cvt_pkrtz(p1[2*r], p1[2*r+1]));
        }
        union { int i[8]; v16h v; } bPu;
#pragma unroll
        for (int i = 0; i < 4; ++i) {
            int send = hi ? Ad[i] : Bd[i];
            int keep = hi ? Bd[i] : Ad[i];
            int recv = __shfl_xor(send, 16, 32);
            bPu.i[i]     = hi ? recv : keep;   // halves 0..7  : m0+16*hi+0..7
            bPu.i[i + 4] = hi ? keep : recv;   // halves 8..15 : m0+16*hi+8..15
        }

        // rescale accumulators by exp2(m_old - m_new)
#pragma unroll
        for (int r = 0; r < 8; ++r) { acc0[r] *= fac; acc1[r] *= fac; }

        // A operand (V^T rows): lane row d = col (tile0) / 16+col (tile1)
        v16h aV0, aV1;
        {
            const _Float16* v0 = vbase + (size_t)col * Nn + m0;
            const _Float16* v1 = vbase + (size_t)(16 + col) * Nn + m0;
#pragma unroll
            for (int j = 0; j < 8; ++j) {
                aV0[j]     = v0[16 * hi + j];
                aV0[j + 8] = v0[16 * hi + 8 + j];
                aV1[j]     = v1[16 * hi + j];
                aV1[j + 8] = v1[16 * hi + 8 + j];
            }
        }
        acc0 = __builtin_amdgcn_wmma_f32_16x16x32_f16(false, aV0, false, bPu.v,
                                                      (short)0, acc0, false, false);
        acc1 = __builtin_amdgcn_wmma_f32_16x16x32_f16(false, aV1, false, bPu.v,
                                                      (short)0, acc1, false, false);
    }

    float ltot = lsum + __shfl_xor(lsum, 16, 32);
    float inv  = 1.0f / ltot;

    // out[b][q][h*32 + d], lane has q = qtile*16+col, d = {r+8hi} u {16+r+8hi}
    float* orow = out + ((size_t)b * Nn + (qtile * 16 + col)) * Dd + h * DHh;
    float4 o;
    o = make_float4(acc0[0]*inv, acc0[1]*inv, acc0[2]*inv, acc0[3]*inv);
    *(float4*)(orow + 8 * hi) = o;
    o = make_float4(acc0[4]*inv, acc0[5]*inv, acc0[6]*inv, acc0[7]*inv);
    *(float4*)(orow + 8 * hi + 4) = o;
    o = make_float4(acc1[0]*inv, acc1[1]*inv, acc1[2]*inv, acc1[3]*inv);
    *(float4*)(orow + 16 + 8 * hi) = o;
    o = make_float4(acc1[4]*inv, acc1[5]*inv, acc1[6]*inv, acc1[7]*inv);
    *(float4*)(orow + 16 + 8 * hi + 4) = o;
}

extern "C" void kernel_launch(void* const* d_in, const int* in_sizes, int n_in,
                              void* d_out, int out_size, void* d_ws, size_t ws_size,
                              hipStream_t stream)
{
    const float* x  = (const float*)d_in[0];
    const float* Wq = (const float*)d_in[1];
    const float* bq = (const float*)d_in[2];
    const float* Wk = (const float*)d_in[3];
    const float* bk = (const float*)d_in[4];
    const float* Wv = (const float*)d_in[5];
    const float* bv = (const float*)d_in[6];
    float* out = (float*)d_out;

    const size_t perTensor = (size_t)Bb * Hh * Nn * DHh;   // 2M f16 elems = 4MB
    _Float16* Qh = (_Float16*)d_ws;
    _Float16* Kh = Qh + perTensor;
    _Float16* Vt = Kh + perTensor;

    const int nElem = Bb * Nn * Hh * DHh;                  // 2,097,152
    qkv_proj<<<nElem / 256, 256, 0, stream>>>(x, Wq, bq, Wk, bk, Wv, bv, Qh, Kh, Vt);

    const int nTiles = Bb * Hh * (Nn / 16);                // 4096 waves
    attn<<<nTiles / 4, 128, 0, stream>>>(Qh, Kh, Vt, out);
}
